// Head_75840532513144
// MI455X (gfx1250) — compile-verified, matrix operand order checked by reading, not run on hardware
//
#include <hip/hip_runtime.h>
#include <hip/hip_bf16.h>

#define B_DIM 512
#define T_DIM 256
#define C_DIM 256
#define H_DIM 64
#define SCALE 0.0625f   // C^-0.5 = 256^-0.5

typedef __attribute__((ext_vector_type(16))) __bf16 bf16x16;
typedef __attribute__((ext_vector_type(8)))  float  f32x8;

union BF16x16U { bf16x16 v; uint4 u[2]; __bf16 h[16]; };

__device__ __forceinline__ f32x8 f32x8_zero() {
  f32x8 z = {0.f,0.f,0.f,0.f,0.f,0.f,0.f,0.f};
  return z;
}

__device__ __forceinline__ f32x8 wmma_bf16(bf16x16 a, bf16x16 b, f32x8 c) {
  // D = A(16x32) * B(32x16) + C, bf16 in / f32 out
  return __builtin_amdgcn_wmma_f32_16x16x32_bf16(false, a, false, b, (short)0, c,
                                                 false, false);
}

// Load a WMMA A/B operand from row-major bf16 LDS.
// Caller passes p = &row[col0 + o0]; halves 0..7 = cols +0..7, halves 8..15 = cols +16..23.
__device__ __forceinline__ bf16x16 lds_ld_ab(const __bf16* p) {
  BF16x16U t;
  t.u[0] = *reinterpret_cast<const uint4*>(p);
  t.u[1] = *reinterpret_cast<const uint4*>(p + 16);
  return t.v;
}

// Load a WMMA A operand from row-major fp32 global memory, converting to bf16.
__device__ __forceinline__ bf16x16 gld_x_a(const float* p) {
  float4 f0 = *reinterpret_cast<const float4*>(p);
  float4 f1 = *reinterpret_cast<const float4*>(p + 4);
  float4 f2 = *reinterpret_cast<const float4*>(p + 16);
  float4 f3 = *reinterpret_cast<const float4*>(p + 20);
  BF16x16U t;
  t.h[0]=(__bf16)f0.x; t.h[1]=(__bf16)f0.y; t.h[2]=(__bf16)f0.z; t.h[3]=(__bf16)f0.w;
  t.h[4]=(__bf16)f1.x; t.h[5]=(__bf16)f1.y; t.h[6]=(__bf16)f1.z; t.h[7]=(__bf16)f1.w;
  t.h[8]=(__bf16)f2.x; t.h[9]=(__bf16)f2.y; t.h[10]=(__bf16)f2.z; t.h[11]=(__bf16)f2.w;
  t.h[12]=(__bf16)f3.x; t.h[13]=(__bf16)f3.y; t.h[14]=(__bf16)f3.z; t.h[15]=(__bf16)f3.w;
  return t.v;
}

// LDS element layout (bf16):
//   WT[3]  : [H][C] transposed weights, 3 * 64*256
//   Qs     : [T][H] 256*64
//   Ks     : [T][H] 256*64
//   VTs    : [H][T] 64*256
//   Pstage : per-wave 16x32 staging for softmax probs, 8 * 512
#define LDS_ELEMS (3*H_DIM*C_DIM + 2*T_DIM*H_DIM + H_DIM*T_DIM + 8*16*32)
#define LDS_BYTES (LDS_ELEMS * 2)

__global__ __launch_bounds__(256, 1)
void head_attn_kernel(const float* __restrict__ x,
                      const float* __restrict__ Wq,
                      const float* __restrict__ Wk,
                      const float* __restrict__ Wv,
                      float* __restrict__ out) {
  extern __shared__ __align__(16) char smem_raw[];
  __bf16* smem = reinterpret_cast<__bf16*>(smem_raw);
  __bf16* WT0 = smem;
  __bf16* WT1 = WT0 + H_DIM*C_DIM;
  __bf16* WT2 = WT1 + H_DIM*C_DIM;
  __bf16* Qs  = WT2 + H_DIM*C_DIM;
  __bf16* Ks  = Qs  + T_DIM*H_DIM;
  __bf16* VTs = Ks  + T_DIM*H_DIM;
  __bf16* Pst = VTs + H_DIM*T_DIM;

  const int tid  = threadIdx.x;
  const int wv   = tid >> 5;        // wave id 0..7
  const int lane = tid & 31;
  const int lr   = lane & 15;       // lane within half
  const int half = lane >> 4;       // 0 or 1
  const int o0   = half * 8;        // K-offset for A/B operand layout
  const int b    = blockIdx.x;

  // ---- Phase 0: stage transposed weights into LDS as bf16 ----
  {
    const float* Ws[3] = {Wq, Wk, Wv};
    __bf16* Ds[3] = {WT0, WT1, WT2};
    #pragma unroll
    for (int m = 0; m < 3; ++m) {
      const float* W = Ws[m];
      __bf16* dst = Ds[m];
      for (int i4 = tid; i4 < (C_DIM*H_DIM)/4; i4 += 256) {
        float4 f = reinterpret_cast<const float4*>(W)[i4];
        int c = (i4*4) / H_DIM;
        int h = (i4*4) % H_DIM;
        dst[(h+0)*C_DIM + c] = (__bf16)f.x;
        dst[(h+1)*C_DIM + c] = (__bf16)f.y;
        dst[(h+2)*C_DIM + c] = (__bf16)f.z;
        dst[(h+3)*C_DIM + c] = (__bf16)f.w;
      }
    }
  }
  __syncthreads();

  // ---- Phase 1: Q/K/V projections, each wave owns T-tiles wv and wv+8 ----
  for (int tt = wv; tt < T_DIM/16; tt += 8) {
    f32x8 qacc[4], kacc[4], vacc[4];
    #pragma unroll
    for (int ht = 0; ht < 4; ++ht) {
      qacc[ht] = f32x8_zero(); kacc[ht] = f32x8_zero(); vacc[ht] = f32x8_zero();
    }
    const float* xrow = x + ((size_t)b*T_DIM + 16*tt + lr) * C_DIM;
    #pragma unroll
    for (int cc = 0; cc < C_DIM/32; ++cc) {
      if (cc + 1 < C_DIM/32)
        __builtin_prefetch(xrow + 32*(cc+1) + o0, 0, 1);   // global_prefetch_b8
      bf16x16 a = gld_x_a(xrow + 32*cc + o0);
      #pragma unroll
      for (int ht = 0; ht < 4; ++ht) {
        bf16x16 bq = lds_ld_ab(WT0 + (16*ht + lr)*C_DIM + 32*cc + o0);
        bf16x16 bk = lds_ld_ab(WT1 + (16*ht + lr)*C_DIM + 32*cc + o0);
        bf16x16 bvv = lds_ld_ab(WT2 + (16*ht + lr)*C_DIM + 32*cc + o0);
        qacc[ht] = wmma_bf16(a, bq, qacc[ht]);
        kacc[ht] = wmma_bf16(a, bk, kacc[ht]);
        vacc[ht] = wmma_bf16(a, bvv, vacc[ht]);
      }
    }
    // D layout: lane holds col n = 16*ht + lr, rows m = 8*half + r
    #pragma unroll
    for (int ht = 0; ht < 4; ++ht) {
      int h = 16*ht + lr;
      #pragma unroll
      for (int r = 0; r < 8; ++r) {
        int t = 16*tt + 8*half + r;
        Qs[t*H_DIM + h]  = (__bf16)qacc[ht][r];
        Ks[t*H_DIM + h]  = (__bf16)kacc[ht][r];
        VTs[h*T_DIM + t] = (__bf16)vacc[ht][r];
      }
    }
  }
  __syncthreads();

  // ---- Phase 2: causal flash attention, each wave owns query tiles wv, wv+8 ----
  __bf16* myP = Pst + wv * (16*32);
  for (int qt = wv; qt < T_DIM/16; qt += 8) {
    bf16x16 aQ[2];
    #pragma unroll
    for (int hc = 0; hc < 2; ++hc)
      aQ[hc] = lds_ld_ab(Qs + (16*qt + lr)*H_DIM + 32*hc + o0);

    f32x8 oacc[4];
    #pragma unroll
    for (int ht = 0; ht < 4; ++ht) oacc[ht] = f32x8_zero();
    float mrun[8], lrun[8];
    #pragma unroll
    for (int r = 0; r < 8; ++r) { mrun[r] = -1e30f; lrun[r] = 0.f; }

    const int nchunks = (16*qt + 16 + 31) / 32;   // causal: only needed key chunks
    for (int j = 0; j < nchunks; ++j) {
      // S tile: queries [16qt,16qt+16) x keys [32j,32j+32)
      f32x8 s0 = f32x8_zero(), s1 = f32x8_zero();
      #pragma unroll
      for (int hc = 0; hc < 2; ++hc) {
        bf16x16 bk0 = lds_ld_ab(Ks + (32*j + lr)*H_DIM + 32*hc + o0);
        bf16x16 bk1 = lds_ld_ab(Ks + (32*j + 16 + lr)*H_DIM + 32*hc + o0);
        s0 = wmma_bf16(aQ[hc], bk0, s0);
        s1 = wmma_bf16(aQ[hc], bk1, s1);
      }
      // scale + causal mask (per-lane select, no divergence)
      const int k0 = 32*j + lr, k1 = 32*j + 16 + lr;
      float p0[8], p1[8], rm[8];
      #pragma unroll
      for (int r = 0; r < 8; ++r) {
        int tq = 16*qt + 8*half + r;
        float v0 = s0[r]*SCALE; if (k0 > tq) v0 = -1e30f;
        float v1 = s1[r]*SCALE; if (k1 > tq) v1 = -1e30f;
        p0[r] = v0; p1[r] = v1;
        rm[r] = fmaxf(v0, v1);
      }
      // row max across the 16 lanes of each half (rows m=8*half+r)
      #pragma unroll
      for (int off = 1; off < 16; off <<= 1)
        #pragma unroll
        for (int r = 0; r < 8; ++r)
          rm[r] = fmaxf(rm[r], __shfl_xor(rm[r], off, 32));
      float alpha[8], rs[8];
      #pragma unroll
      for (int r = 0; r < 8; ++r) {
        float nm = fmaxf(mrun[r], rm[r]);
        alpha[r] = __expf(mrun[r] - nm);
        mrun[r]  = nm;
        p0[r] = __expf(p0[r] - nm);
        p1[r] = __expf(p1[r] - nm);
        rs[r] = p0[r] + p1[r];
      }
      #pragma unroll
      for (int off = 1; off < 16; off <<= 1)
        #pragma unroll
        for (int r = 0; r < 8; ++r)
          rs[r] += __shfl_xor(rs[r], off, 32);
      #pragma unroll
      for (int r = 0; r < 8; ++r) lrun[r] = alpha[r]*lrun[r] + rs[r];
      #pragma unroll
      for (int ht = 0; ht < 4; ++ht)
        #pragma unroll
        for (int r = 0; r < 8; ++r) oacc[ht][r] *= alpha[r];

      // D-layout -> A-layout for P via per-wave LDS staging (16x32 bf16 row-major)
      #pragma unroll
      for (int r = 0; r < 8; ++r) {
        int m = 8*half + r;
        myP[m*32 + lr]      = (__bf16)p0[r];
        myP[m*32 + 16 + lr] = (__bf16)p1[r];
      }
      asm volatile("s_wait_dscnt 0" ::: "memory");  // DS in-order per wave; wait writes done
      bf16x16 aP = lds_ld_ab(myP + lr*32 + o0);
      #pragma unroll
      for (int ht = 0; ht < 4; ++ht) {
        bf16x16 bv = lds_ld_ab(VTs + (16*ht + lr)*T_DIM + 32*j + o0);
        oacc[ht] = wmma_bf16(aP, bv, oacc[ht]);
      }
    }
    // normalize + store (lanes lr=0..15 write 64B contiguous per (ht,r))
    #pragma unroll
    for (int ht = 0; ht < 4; ++ht) {
      int h = 16*ht + lr;
      #pragma unroll
      for (int r = 0; r < 8; ++r) {
        int t = 16*qt + 8*half + r;
        out[((size_t)b*T_DIM + t)*H_DIM + h] = oacc[ht][r] / lrun[r];
      }
    }
  }
}

extern "C" void kernel_launch(void* const* d_in, const int* in_sizes, int n_in,
                              void* d_out, int out_size, void* d_ws, size_t ws_size,
                              hipStream_t stream) {
  (void)in_sizes; (void)n_in; (void)d_ws; (void)ws_size; (void)out_size;
  const float* x  = (const float*)d_in[0];
  const float* Wq = (const float*)d_in[1];
  const float* Wk = (const float*)d_in[2];
  const float* Wv = (const float*)d_in[3];
  float* out = (float*)d_out;

  // 200 KB dynamic LDS (within the 320 KB WGP budget); idempotent, capture-safe.
  hipFuncSetAttribute((const void*)head_attn_kernel,
                      hipFuncAttributeMaxDynamicSharedMemorySize, LDS_BYTES);

  head_attn_kernel<<<B_DIM, 256, LDS_BYTES, stream>>>(x, Wq, Wk, Wv, out);
}
